// Block_35682588295597
// MI455X (gfx1250) — matrix operand order repaired and checked by reading, not compiled, not run on hardware
//
#include <hip/hip_runtime.h>
#include <hip/hip_bf16.h>
#include <math.h>

typedef __attribute__((ext_vector_type(16))) _Float16 v16h;
typedef __attribute__((ext_vector_type(8)))  _Float16 v8h;
typedef __attribute__((ext_vector_type(8)))  float    v8f;

#define C_DIM   256
#define T_DIM   64
#define V_DIM   25
#define L_DIM   125
#define LPAD    128
#define N_DIM   32
#define NBLK    (N_DIM * T_DIM)        // 2048 workgroups
#define NTHREADS 512

#define KT_TILES 8                     // 256 / 32
#define NT_TILES 16                    // 256 / 16
#define W_HALFS  (KT_TILES * NT_TILES * 32 * 16)   // 65536 halfs per packed weight

// packed weight offsets (in halfs) inside wpack
#define WQ_OFF  (0 * W_HALFS)
#define WK_OFF  (1 * W_HALFS)
#define WV_OFF  (2 * W_HALFS)
#define WT_OFF  (3 * W_HALFS)
#define WP_OFF  (4 * W_HALFS)
#define W1_OFF  (5 * W_HALFS)
#define W2_OFF  (6 * W_HALFS)

// LDS layout (byte offsets)
#define SLOT_BYTES (LPAD * C_DIM * 2)          // 65536 per fp16 slot
#define LOG_OFF    (4 * SLOT_BYTES)            // 262144 : float[128*8]
#define PQC_OFF    (LOG_OFF + 4096)            // float[256]
#define PKC_OFF    (PQC_OFF + 1024)            // float[256]
#define ATTM_OFF   (PKC_OFF + 1024)            // float[32*256] (also qa/ka cache)
#define SMEM_BYTES (ATTM_OFF + 32 * C_DIM * 4) // 301056 bytes (<320KB WGP LDS)

#define XT_ELEMS  (N_DIM * T_DIM * V_DIM * C_DIM)   // 13,107,200 floats
#define XT_BYTES  ((size_t)XT_ELEMS * 4)

// -------------------- prep kernels --------------------

// x (N,C,T,V) -> xt (N,T,V,C)  (coalesced writes)
__global__ void prep_x(const float* __restrict__ x, float* __restrict__ xt) {
    int idx = blockIdx.x * 256 + threadIdx.x;        // idx over xt layout
    int c  = idx & 255;
    int r  = idx >> 8;          // (n*64 + t)*25 + v
    int v  = r % 25;
    int r2 = r / 25;            // n*64 + t
    int t  = r2 & 63;
    int n  = r2 >> 6;
    xt[idx] = x[((n * C_DIM + c) * T_DIM + t) * V_DIM + v];
}

// Pack seven 256x256 fp32 weights (row-major [in=K][out=N]) into wave32 WMMA
// B-fragment order: for tile (kt,nt), lane holds 16 contiguous halfs:
//   lane<16  -> column nt*16+lane,      K = kt*32 + {0..7,16..23}
//   lane>=16 -> column nt*16+(lane-16), K = kt*32 + {8..15,24..31}
__global__ void prep_w(const float* __restrict__ qw, const float* __restrict__ kw,
                       const float* __restrict__ vw, const float* __restrict__ tw,
                       const float* __restrict__ pw, const float* __restrict__ w1,
                       const float* __restrict__ w2, _Float16* __restrict__ wpack) {
    const float* srcs[7] = {qw, kw, vw, tw, pw, w1, w2};
    int wIdx = blockIdx.x >> 7;          // 0..6
    int r    = blockIdx.x & 127;         // kt*16 + nt
    int kt = r >> 4, nt = r & 15;
    int lane = threadIdx.x;              // 32 threads
    const float* W = srcs[wIdx];
    _Float16* dst = wpack + wIdx * W_HALFS + (((kt * 16 + nt) * 32 + lane) << 4);
    int kb = kt * 32 + (lane >> 4) * 8;
    int n  = nt * 16 + (lane & 15);
#pragma unroll
    for (int i = 0; i < 16; ++i) dst[i] = (_Float16)W[(kb + i + (i >= 8 ? 8 : 0)) * C_DIM + n];
}

// -------------------- WMMA fragment helpers --------------------

// A fragment (16x32 fp16) from LDS row-major [rows x 256].
// ISA layout: lane m (<16) row m, K {0..7,16..23}; lane m+16 row m, K {8..15,24..31}.
__device__ __forceinline__ v16h ldA(const _Float16* A, int strip, int kt, int lane) {
    const _Float16* p = A + (strip * 16 + (lane & 15)) * C_DIM + kt * 32 + ((lane >> 4) * 8);
    v8h a0 = *(const v8h*)p;
    v8h a1 = *(const v8h*)(p + 16);
    return __builtin_shufflevector(a0, a1, 0,1,2,3,4,5,6,7,8,9,10,11,12,13,14,15);
}

// B fragment (32x16 fp16) from packed global weight (see prep_w).
__device__ __forceinline__ v16h ldB(const _Float16* B, int kt, int nt, int lane) {
    const _Float16* p = B + (((kt * 16 + nt) * 32 + lane) << 4);
    v8h b0 = *(const v8h*)p;
    v8h b1 = *(const v8h*)(p + 8);
    return __builtin_shufflevector(b0, b1, 0,1,2,3,4,5,6,7,8,9,10,11,12,13,14,15);
}

// Hoist all 8 A fragments for one 16-row strip (64 VGPRs, reused across nt).
__device__ __forceinline__ void loadA8(const _Float16* A, int strip, int lane, v16h af[8]) {
#pragma unroll
    for (int kt = 0; kt < KT_TILES; ++kt) af[kt] = ldA(A, strip, kt, lane);
}

// One 16x16 output tile from preloaded A fragments (8 WMMAs, K=256).
__device__ __forceinline__ v8f tileMMpre(const v16h af[8], const _Float16* B,
                                         int nt, int lane) {
    v8f acc = {};
#pragma unroll
    for (int kt = 0; kt < KT_TILES; ++kt)
        acc = __builtin_amdgcn_wmma_f32_16x16x32_f16(false, af[kt], false,
                                                     ldB(B, kt, nt, lane),
                                                     (short)0, acc, false, false);
    return acc;
}

// 128x256 = A(128x256,LDS fp16) * W(256x256,packed) + bias -> D (LDS fp16)
// 16 waves: wave w -> strip w>>1, nt range (w&1)*8 .. +8 ; A fragments hoisted.
__device__ __forceinline__ void gemmBiasF16(const _Float16* A, const _Float16* B,
                                            const float* __restrict__ bias,
                                            _Float16* D, int wave, int lane) {
    int strip = wave >> 1;
    int ntBase = (wave & 1) * 8;
    v16h af[8];
    loadA8(A, strip, lane, af);
    for (int q = 0; q < 8; ++q) {
        int nt = ntBase + q;
        v8f acc = tileMMpre(af, B, nt, lane);
        int col  = nt * 16 + (lane & 15);
        int row0 = strip * 16 + ((lane >> 4) * 8);
        float bb = bias[col];
#pragma unroll
        for (int j = 0; j < 8; ++j)
            D[(row0 + j) * C_DIM + col] = (_Float16)(acc[j] + bb);
    }
}

// -------------------- main fused kernel --------------------

__global__ void __launch_bounds__(NTHREADS)
fused_block(const float* __restrict__ xt, const _Float16* __restrict__ wpack,
            const float* __restrict__ ln1_g, const float* __restrict__ ln1_b,
            const float* __restrict__ q_b,
            const float* __restrict__ qa_w, const float* __restrict__ qa_b,
            const float* __restrict__ k_b,
            const float* __restrict__ ka_w, const float* __restrict__ ka_b,
            const float* __restrict__ v_b,  const float* __restrict__ t_b,
            const float* __restrict__ p_b,
            const float* __restrict__ ffn_g, const float* __restrict__ ffn_b,
            const float* __restrict__ b1,   const float* __restrict__ b2,
            float* __restrict__ out) {
    extern __shared__ char smem[];
    _Float16* s0 = (_Float16*)smem;                       // nx -> kv -> attin
    _Float16* s1 = (_Float16*)(smem + 1 * SLOT_BYTES);    // q (alive to end) -> LN2 tile
    _Float16* s2 = (_Float16*)(smem + 2 * SLOT_BYTES);    // k -> att
    _Float16* s3 = (_Float16*)(smem + 3 * SLOT_BYTES);    // v -> kvh -> gelu(h1)
    float* slog  = (float*)(smem + LOG_OFF);              // [128*8]
    float* spqc  = (float*)(smem + PQC_OFF);              // [256]
    float* spkc  = (float*)(smem + PKC_OFF);              // [256]
    float* sattm = (float*)(smem + ATTM_OFF);             // [32*256] / qa,ka cache

    const int tid  = threadIdx.x;
    const int lane = tid & 31;
    const int wave = tid >> 5;
    const int b = blockIdx.x;
    const int n = b >> 6, t = b & 63;
    const float scale = 0.17677669529663687f;             // 1/sqrt(32)

    // ---- Phase 1: gather window rows + LayerNorm -> nx (fp16, s0) ----
    for (int i = 0; i < 8; ++i) {
        int l = wave * 8 + i;                             // 0..127
        int c0 = lane * 8;
        float vals[8];
        const float* src = nullptr;
        if (l < L_DIM) {
            int w = l / 25, vv = l - w * 25;
            int tt = t + w - 2;
            if (tt >= 0 && tt < T_DIM) src = xt + ((((n * T_DIM + tt) * V_DIM) + vv) << 8);
        }
        float s = 0.f, ss = 0.f;
        if (src) {
            float4 f0 = *(const float4*)(src + c0);
            float4 f1 = *(const float4*)(src + c0 + 4);
            vals[0]=f0.x; vals[1]=f0.y; vals[2]=f0.z; vals[3]=f0.w;
            vals[4]=f1.x; vals[5]=f1.y; vals[6]=f1.z; vals[7]=f1.w;
#pragma unroll
            for (int u = 0; u < 8; ++u) { s += vals[u]; ss += vals[u]*vals[u]; }
        } else {
#pragma unroll
            for (int u = 0; u < 8; ++u) vals[u] = 0.f;
        }
        for (int off = 16; off; off >>= 1) { s += __shfl_xor(s, off); ss += __shfl_xor(ss, off); }
        float m = s * (1.f / 256.f);
        float rstd = rsqrtf(ss * (1.f / 256.f) - m * m + 1e-5f);
        v8h nv;
#pragma unroll
        for (int u = 0; u < 8; ++u) {
            int c = c0 + u;
            float z = (l < L_DIM) ? ((vals[u] - m) * rstd * ln1_g[c] + ln1_b[c]) : 0.f;
            nv[u] = (_Float16)z;
        }
        *(v8h*)(s0 + l * C_DIM + c0) = nv;
    }
    __syncthreads();

    // ---- Phase 2: q,k,v projections (WMMA) ----
    gemmBiasF16(s0, wpack + WQ_OFF, q_b, s1, wave, lane);
    gemmBiasF16(s0, wpack + WK_OFF, k_b, s2, wave, lane);
    gemmBiasF16(s0, wpack + WV_OFF, v_b, s3, wave, lane);
    // cache qa_w transposed into sattm: qaT[h*256+c] = qa_w[c*8+h]
    for (int i = tid; i < 8 * C_DIM; i += NTHREADS) {
        int h = i >> 8, c = i & 255;
        sattm[i] = qa_w[c * 8 + h];
    }
    __syncthreads();

    // ---- Phase 3: q softmax-pool -> pqc[256] ----
    for (int i = tid; i < L_DIM * 8; i += NTHREADS) {
        int l = i >> 3, h = i & 7;
        const _Float16* row = s1 + l * C_DIM;
        const float* wrow = sattm + h * C_DIM;
        float acc = 0.f;
#pragma unroll 4
        for (int c = 0; c < C_DIM; c += 8) {
            v8h a = *(const v8h*)(row + c);
            float4 w0 = *(const float4*)(wrow + c);
            float4 w1 = *(const float4*)(wrow + c + 4);
            acc += (float)a[0]*w0.x + (float)a[1]*w0.y + (float)a[2]*w0.z + (float)a[3]*w0.w
                 + (float)a[4]*w1.x + (float)a[5]*w1.y + (float)a[6]*w1.z + (float)a[7]*w1.w;
        }
        slog[l * 8 + h] = (acc + qa_b[h]) * scale;
    }
    __syncthreads();
    if (wave < 8) {                                       // softmax over L per head
        int h = wave;
        float mx = -1e30f;
        for (int l = lane; l < L_DIM; l += 32) mx = fmaxf(mx, slog[l * 8 + h]);
        for (int off = 16; off; off >>= 1) mx = fmaxf(mx, __shfl_xor(mx, off));
        float s = 0.f;
        for (int l = lane; l < L_DIM; l += 32) {
            float e = __expf(slog[l * 8 + h] - mx); slog[l * 8 + h] = e; s += e;
        }
        for (int off = 16; off; off >>= 1) s += __shfl_xor(s, off);
        float inv = 1.f / s;
        for (int l = lane; l < L_DIM; l += 32) slog[l * 8 + h] *= inv;
    }
    __syncthreads();
    if (tid < C_DIM) {
        int h = tid >> 5;
        float acc = 0.f;
        for (int l = 0; l < L_DIM; ++l) acc += slog[l * 8 + h] * (float)s1[l * C_DIM + tid];
        spqc[tid] = acc;
    }
    __syncthreads();
    // re-fill cache with ka_w transposed
    for (int i = tid; i < 8 * C_DIM; i += NTHREADS) {
        int h = i >> 8, c = i & 255;
        sattm[i] = ka_w[c * 8 + h];
    }
    __syncthreads();

    // ---- Phase 4: k softmax-pool (on k*pq) -> pkc[256] ----
    for (int i = tid; i < L_DIM * 8; i += NTHREADS) {
        int l = i >> 3, h = i & 7;
        const _Float16* row = s2 + l * C_DIM;
        const float* wrow = sattm + h * C_DIM;
        float acc = 0.f;
#pragma unroll 4
        for (int c = 0; c < C_DIM; c += 8) {
            v8h a = *(const v8h*)(row + c);
            float4 w0 = *(const float4*)(wrow + c);
            float4 w1 = *(const float4*)(wrow + c + 4);
            float4 p0 = *(const float4*)(spqc + c);
            float4 p1 = *(const float4*)(spqc + c + 4);
            acc += (float)a[0]*p0.x*w0.x + (float)a[1]*p0.y*w0.y
                 + (float)a[2]*p0.z*w0.z + (float)a[3]*p0.w*w0.w
                 + (float)a[4]*p1.x*w1.x + (float)a[5]*p1.y*w1.y
                 + (float)a[6]*p1.z*w1.z + (float)a[7]*p1.w*w1.w;
        }
        slog[l * 8 + h] = (acc + ka_b[h]) * scale;
    }
    __syncthreads();
    if (wave < 8) {
        int h = wave;
        float mx = -1e30f;
        for (int l = lane; l < L_DIM; l += 32) mx = fmaxf(mx, slog[l * 8 + h]);
        for (int off = 16; off; off >>= 1) mx = fmaxf(mx, __shfl_xor(mx, off));
        float s = 0.f;
        for (int l = lane; l < L_DIM; l += 32) {
            float e = __expf(slog[l * 8 + h] - mx); slog[l * 8 + h] = e; s += e;
        }
        for (int off = 16; off; off >>= 1) s += __shfl_xor(s, off);
        float inv = 1.f / s;
        for (int l = lane; l < L_DIM; l += 32) slog[l * 8 + h] *= inv;
    }
    __syncthreads();
    if (tid < C_DIM) {
        int h = tid >> 5;
        float acc = 0.f;
        for (int l = 0; l < L_DIM; ++l) acc += slog[l * 8 + h] * (float)s2[l * C_DIM + tid];
        spkc[tid] = acc;
    }
    __syncthreads();

    // ---- Phase 5: kv = v * pk  (fp16 -> s0; rows>=125 stay zero) ----
    for (int i = tid * 8; i < L_DIM * C_DIM; i += NTHREADS * 8) {
        v8h v = *(const v8h*)(s3 + i);
        int c = i & 255;
        float4 p0 = *(const float4*)(spkc + c);
        float4 p1 = *(const float4*)(spkc + c + 4);
        v8h r;
        r[0]=(_Float16)((float)v[0]*p0.x); r[1]=(_Float16)((float)v[1]*p0.y);
        r[2]=(_Float16)((float)v[2]*p0.z); r[3]=(_Float16)((float)v[3]*p0.w);
        r[4]=(_Float16)((float)v[4]*p1.x); r[5]=(_Float16)((float)v[5]*p1.y);
        r[6]=(_Float16)((float)v[6]*p1.z); r[7]=(_Float16)((float)v[7]*p1.w);
        *(v8h*)(s0 + i) = r;
    }
    __syncthreads();

    // ---- Phase 6: kvh = kv @ t_w + t_b -> s3 ----
    gemmBiasF16(s0, wpack + WT_OFF, t_b, s3, wave, lane);
    __syncthreads();

    // ---- Phase 7: attin = kvh + q -> s0 ----
    for (int i = tid * 8; i < L_DIM * C_DIM; i += NTHREADS * 8) {
        v8h a = *(const v8h*)(s3 + i);
        v8h q = *(const v8h*)(s1 + i);
        v8h r;
#pragma unroll
        for (int u = 0; u < 8; ++u) r[u] = (_Float16)((float)a[u] + (float)q[u]);
        *(v8h*)(s0 + i) = r;
    }
    __syncthreads();

    // ---- Phase 8: att = attin @ p_w + p_b -> s2 ----
    gemmBiasF16(s0, wpack + WP_OFF, p_b, s2, wave, lane);
    __syncthreads();

    // ---- Phase 9: window mean + residual with x -> sattm (fp32, 25x256) ----
    for (int i = tid * 8; i < V_DIM * C_DIM; i += NTHREADS * 8) {
        int vv = i >> 8, c = i & 255;
        float acc[8] = {0.f,0.f,0.f,0.f,0.f,0.f,0.f,0.f};
#pragma unroll
        for (int w = 0; w < 5; ++w) {
            v8h a = *(const v8h*)(s2 + (w * 25 + vv) * C_DIM + c);
#pragma unroll
            for (int u = 0; u < 8; ++u) acc[u] += (float)a[u];
        }
        const float* xr = xt + ((((n * T_DIM + t) * V_DIM) + vv) << 8) + c;
        float4 x0 = *(const float4*)xr;
        float4 x1 = *(const float4*)(xr + 4);
        float4 r0, r1;
        r0.x = acc[0]*0.2f + x0.x; r0.y = acc[1]*0.2f + x0.y;
        r0.z = acc[2]*0.2f + x0.z; r0.w = acc[3]*0.2f + x0.w;
        r1.x = acc[4]*0.2f + x1.x; r1.y = acc[5]*0.2f + x1.y;
        r1.z = acc[6]*0.2f + x1.z; r1.w = acc[7]*0.2f + x1.w;
        *(float4*)(sattm + i) = r0;
        *(float4*)(sattm + i + 4) = r1;
    }
    __syncthreads();

    // ---- Phase 10: LN2 -> fp16 A-tile (32x256) in s1 ----
    for (int i = 0; i < 2; ++i) {
        int r = wave * 2 + i;                  // 0..31
        int c0 = lane * 8;
        bool act = (r < V_DIM);
        float vals[8]; float s = 0.f, ss = 0.f;
        if (act) {
            float4 f0 = *(const float4*)(sattm + r * C_DIM + c0);
            float4 f1 = *(const float4*)(sattm + r * C_DIM + c0 + 4);
            vals[0]=f0.x; vals[1]=f0.y; vals[2]=f0.z; vals[3]=f0.w;
            vals[4]=f1.x; vals[5]=f1.y; vals[6]=f1.z; vals[7]=f1.w;
#pragma unroll
            for (int u = 0; u < 8; ++u) { s += vals[u]; ss += vals[u]*vals[u]; }
        } else {
#pragma unroll
            for (int u = 0; u < 8; ++u) vals[u] = 0.f;
        }
        for (int off = 16; off; off >>= 1) { s += __shfl_xor(s, off); ss += __shfl_xor(ss, off); }
        float m = s * (1.f / 256.f);
        float rstd = rsqrtf(ss * (1.f / 256.f) - m * m + 1e-5f);
        v8h nv;
#pragma unroll
        for (int u = 0; u < 8; ++u) {
            int c = c0 + u;
            float z = act ? ((vals[u] - m) * rstd * ffn_g[c] + ffn_b[c]) : 0.f;
            nv[u] = (_Float16)z;
        }
        *(v8h*)(s1 + r * C_DIM + c0) = nv;
    }
    __syncthreads();

    // ---- Phase 11: FFN layer 1 (gelu) -> s3 (32x256 fp16) ----
    {
        int strip = wave >> 3;                 // both tiles of this wave share a strip
        v16h af[8];
        loadA8(s1, strip, lane, af);
        for (int i = 0; i < 2; ++i) {
            int nt = (wave * 2 + i) & 15;
            v8f acc = tileMMpre(af, wpack + W1_OFF, nt, lane);
            int col  = nt * 16 + (lane & 15);
            int row0 = strip * 16 + ((lane >> 4) * 8);
            float bb = b1[col];
#pragma unroll
            for (int j = 0; j < 8; ++j) {
                float x_ = acc[j] + bb;
                float g = 0.5f * x_ * (1.f + erff(x_ * 0.70710678118f));
                s3[(row0 + j) * C_DIM + col] = (_Float16)g;
            }
        }
    }
    __syncthreads();

    // ---- Phase 12: FFN layer 2 + residual, write out (N,C,T,V) ----
    {
        int strip = wave >> 3;
        v16h af[8];
        loadA8(s3, strip, lane, af);
        for (int i = 0; i < 2; ++i) {
            int nt = (wave * 2 + i) & 15;
            v8f acc = tileMMpre(af, wpack + W2_OFF, nt, lane);
            int col  = nt * 16 + (lane & 15);
            int row0 = strip * 16 + ((lane >> 4) * 8);
            float bb = b2[col];
#pragma unroll
            for (int j = 0; j < 8; ++j) {
                int row = row0 + j;            // = v index
                if (row < V_DIM) {
                    float val = acc[j] + bb + sattm[row * C_DIM + col];
                    out[(((size_t)n * C_DIM + col) * T_DIM + t) * V_DIM + row] = val;
                }
            }
        }
    }
}

// -------------------- host launcher --------------------

extern "C" void kernel_launch(void* const* d_in, const int* in_sizes, int n_in,
                              void* d_out, int out_size, void* d_ws, size_t ws_size,
                              hipStream_t stream) {
    const float* x     = (const float*)d_in[0];
    const float* ln1_g = (const float*)d_in[1];
    const float* ln1_b = (const float*)d_in[2];
    const float* q_w   = (const float*)d_in[3];
    const float* q_b   = (const float*)d_in[4];
    const float* qa_w  = (const float*)d_in[5];
    const float* qa_b  = (const float*)d_in[6];
    const float* k_w   = (const float*)d_in[7];
    const float* k_b   = (const float*)d_in[8];
    const float* ka_w  = (const float*)d_in[9];
    const float* ka_b  = (const float*)d_in[10];
    const float* v_w   = (const float*)d_in[11];
    const float* v_b   = (const float*)d_in[12];
    const float* t_w   = (const float*)d_in[13];
    const float* t_b   = (const float*)d_in[14];
    const float* p_w   = (const float*)d_in[15];
    const float* p_b   = (const float*)d_in[16];
    const float* ffn_g = (const float*)d_in[17];
    const float* ffn_b = (const float*)d_in[18];
    const float* w1    = (const float*)d_in[19];
    const float* b1    = (const float*)d_in[20];
    const float* w2    = (const float*)d_in[21];
    const float* b2    = (const float*)d_in[22];
    float* out = (float*)d_out;

    float*    xt    = (float*)d_ws;
    _Float16* wpack = (_Float16*)((char*)d_ws + XT_BYTES);

    prep_x<<<XT_ELEMS / 256, 256, 0, stream>>>(x, xt);
    prep_w<<<7 * 128, 32, 0, stream>>>(q_w, k_w, v_w, t_w, p_w, w1, w2, wpack);

    hipFuncSetAttribute((const void*)fused_block,
                        hipFuncAttributeMaxDynamicSharedMemorySize, SMEM_BYTES);
    fused_block<<<NBLK, NTHREADS, SMEM_BYTES, stream>>>(
        xt, wpack, ln1_g, ln1_b, q_b, qa_w, qa_b, k_b, ka_w, ka_b,
        v_b, t_b, p_b, ffn_g, ffn_b, b1, b2, out);
}
